// TwoBodyAttn_52295521796553
// MI455X (gfx1250) — matrix operand order repaired and checked by reading, not compiled, not run on hardware
//
#include <hip/hip_runtime.h>
#include <hip/hip_bf16.h>

#define N_TOK 4096
#define HID 256
#define NH 8
#define HD 32

typedef __attribute__((ext_vector_type(16))) __bf16 v16bf;
typedef __attribute__((ext_vector_type(8)))  float  v8f;

union BF16Frag {
    uint4 u[2];
    v16bf v;
};

// 16 contiguous bf16 elements -> one WMMA half-fragment (two b128 loads)
__device__ inline v16bf load16(const __bf16* p) {
    BF16Frag f;
    f.u[0] = *(const uint4*)(p);
    f.u[1] = *(const uint4*)(p + 8);
    return f.v;
}

// A-operand (16x32 bf16, row-major source row of 32 elems):
// lanes 0-15 take K = [0..15], lanes 16-31 take K = [16..31] (contiguous order, as in B)
__device__ inline v16bf load_a32(const __bf16* row, int lhi) {
    BF16Frag f;
    f.u[0] = *(const uint4*)(row + lhi * 16);
    f.u[1] = *(const uint4*)(row + lhi * 16 + 8);
    return f.v;
}

__device__ inline v8f wmma_bf16(v16bf a, v16bf b, v8f c) {
    return __builtin_amdgcn_wmma_f32_16x16x32_bf16(
        /*neg_a=*/false, a, /*neg_b=*/false, b,
        /*c_mod=*/(short)0, c, /*reuse_a=*/false, /*reuse_b=*/false);
}

// ---------------------------------------------------------------------------
// Kernel A: projections (fp32 VALU GEMM, ~1.6 GFLOP) -> head-split layouts
// Q[h][n][d] bf16 (scaled by 32^-0.5), K[h][n][d] bf16, V[h][n][d] f32,
// sg[n][h] = sigmoid(G)
// ---------------------------------------------------------------------------
__global__ __launch_bounds__(256) void proj_kernel(
    const float* __restrict__ feat, const float* __restrict__ Wq,
    const float* __restrict__ bq,   const float* __restrict__ Wkv,
    const float* __restrict__ bkv,  const float* __restrict__ Weg,
    const float* __restrict__ beg,
    __bf16* __restrict__ Qhb, __bf16* __restrict__ Khb,
    float*  __restrict__ Vh,  float* __restrict__ sg)
{
    __shared__ float fsh[8][HID];
    const int t  = threadIdx.x;
    const int r0 = blockIdx.x * 8;

    #pragma unroll
    for (int r = 0; r < 8; ++r)
        fsh[r][t] = feat[(size_t)(r0 + r) * HID + t];
    __syncthreads();

    float qa[8], ka[8], va[8], ga[8];
    #pragma unroll
    for (int r = 0; r < 8; ++r) { qa[r] = 0.f; ka[r] = 0.f; va[r] = 0.f; ga[r] = 0.f; }

    const bool doG = (t < NH);
    #pragma unroll 4
    for (int k = 0; k < HID; ++k) {
        float wq = Wq[(size_t)k * HID + t];
        float wk = Wkv[(size_t)k * (2 * HID) + t];
        float wv = Wkv[(size_t)k * (2 * HID) + HID + t];
        float wg = doG ? Weg[(size_t)k * (2 * NH) + NH + t] : 0.0f;
        #pragma unroll
        for (int r = 0; r < 8; ++r) {
            float f = fsh[r][k];
            qa[r] = fmaf(f, wq, qa[r]);
            ka[r] = fmaf(f, wk, ka[r]);
            va[r] = fmaf(f, wv, va[r]);
            ga[r] = fmaf(f, wg, ga[r]);
        }
    }

    const float qscale = 0.17677669529663687f;  // HEAD_DIM^-0.5
    const int h = t & 7, d = t >> 3;
    const float bqv = bq[t];
    const float bkk = bkv[t];
    const float bkw = bkv[HID + t];
    const float bgv = doG ? beg[NH + t] : 0.f;

    #pragma unroll
    for (int r = 0; r < 8; ++r) {
        int i = r0 + r;
        size_t hd = (size_t)(h * N_TOK + i) * HD + d;
        Qhb[hd] = (__bf16)((qa[r] + bqv) * qscale);
        Khb[hd] = (__bf16)(ka[r] + bkk);
        Vh[hd]  = va[r] + bkw;
        if (doG) {
            float g = ga[r] + bgv;
            sg[(size_t)i * NH + t] = 1.0f / (1.0f + __expf(-g));
        }
    }
}

// ---------------------------------------------------------------------------
// Kernel B: per-(column j, head) softmax stats over i via WMMA score tiles.
// One wave = 16 columns of one head; K^T tile is a loop-invariant B-operand.
// Writes m[h][j] and the gate/denominator-scaled V'^T[h][d][j] (bf16).
// ---------------------------------------------------------------------------
__global__ __launch_bounds__(256) void colstats_kernel(
    const __bf16* __restrict__ Qhb, const __bf16* __restrict__ Khb,
    const float*  __restrict__ Vh,  const float* __restrict__ sg,
    float* __restrict__ mArr, __bf16* __restrict__ VT)
{
    const int lane = threadIdx.x & 31;
    const int wid  = threadIdx.x >> 5;
    const int w    = blockIdx.x * 8 + wid;
    const int h    = w & 7;
    const int j0   = (w >> 3) * 16;
    const int lsub = lane & 15, lhi = lane >> 4;

    // Fixed B operand: K^T tile (lane n: col j0+(n&15), K rows (n>>4)*16..+15)
    const __bf16* krow = Khb + (size_t)(h * N_TOK + j0 + lsub) * HD;
    v16bf bK = load16(krow + lhi * 16);

    const __bf16* qbase = Qhb + (size_t)h * N_TOK * HD;
    float m = -3.0e38f, s = 0.0f;

    for (int it = 0; it < N_TOK / 16; ++it) {
        const __bf16* qrow = qbase + (size_t)(it * 16 + lsub) * HD;
        v16bf aQ = load_a32(qrow, lhi);
        v8f c = {0.f, 0.f, 0.f, 0.f, 0.f, 0.f, 0.f, 0.f};
        c = wmma_bf16(aQ, bK, c);
        // C layout: lane holds 8 rows (i) of one column (j) -> per-lane stats
        float tm = c[0];
        #pragma unroll
        for (int k = 1; k < 8; ++k) tm = fmaxf(tm, c[k]);
        float nm = fmaxf(m, tm);
        float acc = 0.f;
        #pragma unroll
        for (int k = 0; k < 8; ++k) acc += __expf(c[k] - nm);
        s = s * __expf(m - nm) + acc;
        m = nm;
    }
    // merge lane l with lane l+16 (rows 0-7 vs 8-15 of the same column)
    float mo = __shfl_xor(m, 16);
    float so = __shfl_xor(s, 16);
    float mf = fmaxf(m, mo);
    float sf = s * __expf(m - mf) + so * __expf(mo - mf);

    if (lane < 16) mArr[(size_t)h * N_TOK + j0 + lane] = mf;

    // V'[j,d] = V[j,d] * sigmoid(G[j,h]) / denom[j,h], stored transposed:
    // VT[h][d][j] so kernel C's B-operand loads are contiguous in j.
    for (int jj = 0; jj < 16; ++jj) {
        int row = j0 + jj;
        float sj = __shfl(sf, jj);
        float wj = sg[(size_t)row * NH + h] / sj;
        float val = Vh[(size_t)(h * N_TOK + row) * HD + lane] * wj;
        VT[(size_t)(h * HD + lane) * N_TOK + row] = (__bf16)val;
    }
}

// ---------------------------------------------------------------------------
// Kernel C: output pass. One wave = 16 output rows (i) of one head.
// Per 32-j step: 2 score WMMAs -> exp(S - m[j]) -> bf16 P via wave-private
// LDS re-layout -> 2 PV WMMAs accumulating O[16x32] in fp32.
// ---------------------------------------------------------------------------
__global__ __launch_bounds__(256) void attn_out_kernel(
    const __bf16* __restrict__ Qhb, const __bf16* __restrict__ Khb,
    const __bf16* __restrict__ VT,  const float* __restrict__ mArr,
    float* __restrict__ Va)
{
    __shared__ alignas(16) __bf16 Pst[8][16 * 40];  // 40-elem row stride (80B, 16B-mult)
    const int lane = threadIdx.x & 31;
    const int wid  = threadIdx.x >> 5;
    const int w    = blockIdx.x * 8 + wid;
    const int h    = w & 7;
    const int i0   = (w >> 3) * 16;
    const int lsub = lane & 15, lhi = lane >> 4;

    // Fixed A operand: Q tile (rows i0..i0+15, K = head_dim 32)
    const __bf16* qrow = Qhb + (size_t)(h * N_TOK + i0 + lsub) * HD;
    v16bf aQ = load_a32(qrow, lhi);

    const __bf16* kbase  = Khb + (size_t)h * N_TOK * HD;
    const __bf16* vtbase = VT  + (size_t)h * HD * N_TOK;
    const float*  mbase  = mArr + (size_t)h * N_TOK;
    __bf16* pst = &Pst[wid][0];

    v8f O1 = {0.f, 0.f, 0.f, 0.f, 0.f, 0.f, 0.f, 0.f};
    v8f O2 = {0.f, 0.f, 0.f, 0.f, 0.f, 0.f, 0.f, 0.f};

    for (int jt = 0; jt < N_TOK / 32; ++jt) {
        const int j0 = jt * 32;
        // score tiles S[i, j0..j0+15], S[i, j0+16..j0+31]
        v16bf bK1 = load16(kbase + (size_t)(j0 + lsub) * HD + lhi * 16);
        v16bf bK2 = load16(kbase + (size_t)(j0 + 16 + lsub) * HD + lhi * 16);
        v8f S1 = {0.f, 0.f, 0.f, 0.f, 0.f, 0.f, 0.f, 0.f};
        v8f S2 = {0.f, 0.f, 0.f, 0.f, 0.f, 0.f, 0.f, 0.f};
        S1 = wmma_bf16(aQ, bK1, S1);
        S2 = wmma_bf16(aQ, bK2, S2);

        const float m1 = mbase[j0 + lsub];
        const float m2 = mbase[j0 + 16 + lsub];

        // P = exp(S - m[col]) in bf16, staged through wave-private LDS to
        // convert WMMA C-layout -> A-layout. Row = i-offset, col = j-offset.
        const int prow = lhi * 8;
        #pragma unroll
        for (int k = 0; k < 8; ++k) {
            pst[(prow + k) * 40 + lsub]      = (__bf16)__expf(S1[k] - m1);
            pst[(prow + k) * 40 + 16 + lsub] = (__bf16)__expf(S2[k] - m2);
        }
        // LDS ops from one wave are in-order (DScnt); barriers stop the
        // compiler from reordering the cross-lane store->load, wait drains DS.
        __builtin_amdgcn_wave_barrier();
        asm volatile("s_wait_dscnt 0" ::: "memory");
        __builtin_amdgcn_wave_barrier();

        BF16Frag pf;
        pf.u[0] = *(const uint4*)(pst + lsub * 40 + lhi * 16);
        pf.u[1] = *(const uint4*)(pst + lsub * 40 + lhi * 16 + 8);

        // V' tiles: B[k=jj, n=d] = VT[h][d][j0+jj] (contiguous in j)
        v16bf bV1 = load16(vtbase + (size_t)lsub * N_TOK + j0 + lhi * 16);
        v16bf bV2 = load16(vtbase + (size_t)(16 + lsub) * N_TOK + j0 + lhi * 16);
        O1 = wmma_bf16(pf.v, bV1, O1);
        O2 = wmma_bf16(pf.v, bV2, O2);
        asm volatile("" ::: "memory");  // keep next-iter stores after these loads
    }

    // Scatter O back into reference channel order c = d*8 + h
    #pragma unroll
    for (int k = 0; k < 8; ++k) {
        int row = i0 + k + lhi * 8;
        Va[(size_t)row * HID + lsub * 8 + h]        = O1[k];
        Va[(size_t)row * HID + (16 + lsub) * 8 + h] = O2[k];
    }
}

// ---------------------------------------------------------------------------
// Kernel D: LayerNorm (eps = 1e-3) over HIDDEN per row.
// ---------------------------------------------------------------------------
__global__ __launch_bounds__(256) void ln_kernel(
    const float* __restrict__ Va, const float* __restrict__ ln_g,
    const float* __restrict__ ln_b, float* __restrict__ out)
{
    __shared__ float red[16];
    const int i = blockIdx.x, t = threadIdx.x;
    float x = Va[(size_t)i * HID + t];
    float s1 = x, s2 = x * x;
    #pragma unroll
    for (int o = 16; o > 0; o >>= 1) {
        s1 += __shfl_xor(s1, o);
        s2 += __shfl_xor(s2, o);
    }
    const int lane = t & 31, wid = t >> 5;
    if (lane == 0) { red[wid] = s1; red[8 + wid] = s2; }
    __syncthreads();
    float t1 = 0.f, t2 = 0.f;
    #pragma unroll
    for (int k = 0; k < 8; ++k) { t1 += red[k]; t2 += red[8 + k]; }
    float mu  = t1 * (1.0f / HID);
    float var = t2 * (1.0f / HID) - mu * mu;
    out[(size_t)i * HID + t] = (x - mu) * rsqrtf(var + 1e-3f) * ln_g[t] + ln_b[t];
}

// ---------------------------------------------------------------------------
extern "C" void kernel_launch(void* const* d_in, const int* in_sizes, int n_in,
                              void* d_out, int out_size, void* d_ws, size_t ws_size,
                              hipStream_t stream) {
    const float* feat = (const float*)d_in[0];
    const float* Wq   = (const float*)d_in[1];
    const float* bq   = (const float*)d_in[2];
    const float* Wkv  = (const float*)d_in[3];
    const float* bkv  = (const float*)d_in[4];
    const float* Weg  = (const float*)d_in[5];
    const float* beg  = (const float*)d_in[6];
    const float* ln_g = (const float*)d_in[7];
    const float* ln_b = (const float*)d_in[8];
    float* out = (float*)d_out;

    char* ws = (char*)d_ws;
    const size_t MB = 1024 * 1024;
    __bf16* Qhb  = (__bf16*)(ws);               // [8][4096][32] bf16 = 2 MB
    __bf16* Khb  = (__bf16*)(ws + 2 * MB);      // [8][4096][32] bf16 = 2 MB
    float*  Vh   = (float*)(ws + 4 * MB);       // [8][4096][32] f32  = 4 MB
    __bf16* VT   = (__bf16*)(ws + 8 * MB);      // [8][32][4096] bf16 = 2 MB
    float*  sg   = (float*)(ws + 10 * MB);      // [4096][8]     f32  = 128 KB
    float*  mArr = (float*)(ws + 10 * MB + 128 * 1024); // [8][4096] = 128 KB
    float*  Va   = (float*)(ws + 10 * MB + 256 * 1024); // [4096][256] = 4 MB

    proj_kernel<<<N_TOK / 8, 256, 0, stream>>>(feat, Wq, bq, Wkv, bkv, Weg, beg,
                                               Qhb, Khb, Vh, sg);
    colstats_kernel<<<(N_TOK / 16) * NH / 8, 256, 0, stream>>>(Qhb, Khb, Vh, sg,
                                                               mArr, VT);
    attn_out_kernel<<<(N_TOK / 16) * NH / 8, 256, 0, stream>>>(Qhb, Khb, VT, mArr,
                                                               Va);
    ln_kernel<<<N_TOK, 256, 0, stream>>>(Va, ln_g, ln_b, out);
}
